// Model_19885698581241
// MI455X (gfx1250) — matrix operand order repaired and checked by reading, not compiled, not run on hardware
//
#include <hip/hip_runtime.h>
#include <stdint.h>

#define N_ATOMS   65536
#define N_FEAT    512
#define HIDDEN    256
#define N_SPECIES 4
#define N_STRUCT  512

// ---------- types ----------
typedef __bf16   v16bf __attribute__((ext_vector_type(16)));
typedef float    v8f   __attribute__((ext_vector_type(8)));
typedef uint32_t u32x4 __attribute__((ext_vector_type(4), aligned(16)));

union FragU {
    u32x4 u[2];   // 32 bytes
    v16bf v;
};

__device__ __forceinline__ uint16_t f2bf(float f) {
    uint32_t u = __float_as_uint(f);
    uint32_t r = u + 0x7FFFu + ((u >> 16) & 1u);   // round-to-nearest-even
    return (uint16_t)(r >> 16);
}
__device__ __forceinline__ float bf2f(uint16_t h) {
    return __uint_as_float((uint32_t)h << 16);
}

// ---------- setup kernels ----------
__global__ void init_kernel(float* out, int* counts) {
    int t = blockIdx.x * blockDim.x + threadIdx.x;
    if (t < N_STRUCT) out[t] = 0.0f;
    if (t < N_SPECIES) counts[t] = 0;
}

__global__ void partition_kernel(const int* __restrict__ species,
                                 int* counts, int* perm) {
    int i = blockIdx.x * blockDim.x + threadIdx.x;
    if (i >= N_ATOMS) return;
    int s = species[i];
    int slot = atomicAdd(counts + s, 1);
    perm[(size_t)s * N_ATOMS + slot] = i;
}

__global__ void tiles_kernel(const int* counts, int* hdr) {
    if (threadIdx.x == 0 && blockIdx.x == 0) {
        int acc = 0;
        hdr[0] = 0;
        for (int s = 0; s < N_SPECIES; ++s) {
            acc += (counts[s] + 15) >> 4;
            hdr[s + 1] = acc;           // hdr[4] == total tiles
        }
    }
}

// Pre-swizzle f32 weights (K x 256, K = Kt*32) into bf16 WMMA B-fragments.
// Fragment layout (per 16x16 out-tile n, k-block kb): lane L holds 16
// contiguous K values of column n*16 + (L&15), starting at kb*32 + (L>=16)*16.
__global__ void swizzle_w(const float* __restrict__ src, uint16_t* dst, int Kt) {
    int gid = blockIdx.x * blockDim.x + threadIdx.x;
    int total = N_SPECIES * 16 * Kt * 512;
    if (gid >= total) return;
    int q    = gid & 15;
    int L    = (gid >> 4) & 31;
    int kb   = (gid >> 9) % Kt;
    int rest = (gid >> 9) / Kt;
    int n    = rest & 15;
    int sp   = rest >> 4;
    int k    = kb * 32 + ((L >> 4) << 4) + q;
    int c    = (n << 4) + (L & 15);
    int K    = Kt * 32;
    dst[gid] = f2bf(src[((size_t)sp * K + k) * HIDDEN + c]);
}

// ---------- one dense layer: out[16x256] = silu(in[16xK] @ W + b) ----------
template <int KT>
__device__ __forceinline__ void expert_layer(const uint16_t* __restrict__ inLds,
                                             const uint16_t* __restrict__ wfrag,
                                             const float* __restrict__ bias,
                                             uint16_t* __restrict__ outLds,
                                             int lane) {
    const int K    = KT * 32;
    const int r    = lane & 15;
    const int selA = (lane >> 4) * 16;    // contiguous K half-select (matches B fragments)
    const int col  = lane & 15;
    const int mb   = (lane >> 4) * 8;     // accumulator row base

    for (int n = 0; n < 16; ++n) {
        if (n < 15)  // warm L2/L0 for the next N-tile's fragments
            __builtin_prefetch(wfrag + (((size_t)(n + 1) * KT) * 32 + lane) * 16, 0, 1);
        v8f acc = {};
#pragma unroll
        for (int kb = 0; kb < KT; ++kb) {
            FragU fa, fb;
            const uint16_t* abase = inLds + r * K + kb * 32 + selA;
            fa.u[0] = *(const u32x4*)(abase);
            fa.u[1] = *(const u32x4*)(abase + 8);
            const u32x4* pb = (const u32x4*)(wfrag + (((size_t)n * KT + kb) * 32 + lane) * 16);
            fb.u[0] = pb[0];
            fb.u[1] = pb[1];
            acc = __builtin_amdgcn_wmma_f32_16x16x32_bf16(
                false, fa.v, false, fb.v, (short)0, acc, false, false);
        }
        float b = bias[n * 16 + col];
#pragma unroll
        for (int i = 0; i < 8; ++i) {
            float x = acc[i] + b;
            float y = x * __builtin_amdgcn_rcpf(1.0f + __expf(-x));  // SiLU
            outLds[(mb + i) * HIDDEN + n * 16 + col] = f2bf(y);
        }
    }
}

// ---------- main fused MoE-MLP kernel ----------
__global__ void __launch_bounds__(128)
mlp_kernel(const float* __restrict__ features,
           const int* __restrict__ structIdx,
           const float* __restrict__ b1, const float* __restrict__ b2,
           const float* __restrict__ b3, const float* __restrict__ W4,
           const float* __restrict__ b4,
           const int* __restrict__ counts, const int* __restrict__ hdr,
           const int* __restrict__ perm,
           const uint16_t* __restrict__ w1s, const uint16_t* __restrict__ w2s,
           const uint16_t* __restrict__ w3s,
           float* __restrict__ out) {
    extern __shared__ uint16_t smem[];
    const int lane = threadIdx.x & 31;
    const int wib  = threadIdx.x >> 5;
    const int wid  = blockIdx.x * (blockDim.x >> 5) + wib;

    const int totalTiles = hdr[4];
    if (wid >= totalTiles) return;   // whole-wave uniform exit (EXEC all-1s for WMMA)

    int s = 0;
    while (s < 3 && wid >= hdr[s + 1]) ++s;
    const int tb    = (wid - hdr[s]) * 16;
    const int valid = min(16, counts[s] - tb);

    uint16_t* Abuf = smem + wib * (8192 + 4096);  // 16x512 bf16
    uint16_t* Hbuf = Abuf + 8192;                 // 16x256 bf16

    // Stage 16 gathered feature rows into LDS as bf16 (16B global loads/lane).
    const int* plist = perm + (size_t)s * N_ATOMS + tb;
    for (int rr = 0; rr < 16; ++rr) {
        int atom = plist[rr < valid ? rr : (valid - 1)];
        const float4* rowp = (const float4*)(features + (size_t)atom * N_FEAT);
#pragma unroll
        for (int it = 0; it < 4; ++it) {
            float4 v = rowp[lane + it * 32];
            uint32_t p0 = (uint32_t)f2bf(v.x) | ((uint32_t)f2bf(v.y) << 16);
            uint32_t p1 = (uint32_t)f2bf(v.z) | ((uint32_t)f2bf(v.w) << 16);
            uint2* d = (uint2*)(Abuf + rr * N_FEAT + (lane + it * 32) * 4);
            *d = make_uint2(p0, p1);
        }
    }

    // 3 hidden layers on the matrix engine (per-species expert weights).
    expert_layer<16>(Abuf, w1s + (size_t)s * 16 * 16 * 512, b1 + s * HIDDEN, Hbuf, lane);
    expert_layer<8>(Hbuf,  w2s + (size_t)s * 16 * 8  * 512, b2 + s * HIDDEN, Abuf, lane);
    expert_layer<8>(Abuf,  w3s + (size_t)s * 16 * 8  * 512, b3 + s * HIDDEN, Hbuf, lane);

    // Layer 4: 256 -> 1 dot product, split across lane halves + shfl reduce.
    const int r  = lane & 15;
    const int k0 = (lane >> 4) * 128;
    const float* w4 = W4 + s * HIDDEN;
    float sum = 0.0f;
#pragma unroll 8
    for (int i = 0; i < 128; ++i) {
        int k = k0 + i;
        sum += bf2f(Hbuf[r * HIDDEN + k]) * w4[k];
    }
    sum += __shfl_xor(sum, 16, 32);

    // Scatter atomic energies into per-structure bins.
    if (lane < 16 && lane < valid) {
        int atom = plist[lane];
        atomicAdd(out + structIdx[atom], sum + b4[s]);
    }
}

// ---------- host launcher ----------
extern "C" void kernel_launch(void* const* d_in, const int* in_sizes, int n_in,
                              void* d_out, int out_size, void* d_ws, size_t ws_size,
                              hipStream_t stream) {
    const float* features = (const float*)d_in[0];
    const int*   species  = (const int*)d_in[1];
    const int*   structIx = (const int*)d_in[2];
    const float* W1 = (const float*)d_in[3];
    const float* b1 = (const float*)d_in[4];
    const float* W2 = (const float*)d_in[5];
    const float* b2 = (const float*)d_in[6];
    const float* W3 = (const float*)d_in[7];
    const float* b3 = (const float*)d_in[8];
    const float* W4 = (const float*)d_in[9];
    const float* b4 = (const float*)d_in[10];
    float* out = (float*)d_out;

    // workspace layout (bytes): [0)counts 16 | [16)hdr 20 | [4096)perm 1MB |
    // then bf16 fragment weights: W1 1MB, W2 512KB, W3 512KB  (~3 MB total)
    char* ws = (char*)d_ws;
    int* counts = (int*)ws;
    int* hdr    = counts + 4;
    int* perm   = (int*)(ws + 4096);
    uint16_t* w1s = (uint16_t*)(ws + 4096 + (size_t)N_SPECIES * N_ATOMS * 4);
    uint16_t* w2s = w1s + (size_t)N_SPECIES * N_FEAT * HIDDEN;    // +1 MB
    uint16_t* w3s = w2s + (size_t)N_SPECIES * HIDDEN * HIDDEN;    // +512 KB

    init_kernel<<<1, 512, 0, stream>>>(out, counts);
    partition_kernel<<<N_ATOMS / 256, 256, 0, stream>>>(species, counts, perm);
    tiles_kernel<<<1, 32, 0, stream>>>(counts, hdr);

    swizzle_w<<<(N_SPECIES * 16 * 16 * 512 + 255) / 256, 256, 0, stream>>>(W1, w1s, 16);
    swizzle_w<<<(N_SPECIES * 16 * 8  * 512 + 255) / 256, 256, 0, stream>>>(W2, w2s, 8);
    swizzle_w<<<(N_SPECIES * 16 * 8  * 512 + 255) / 256, 256, 0, stream>>>(W3, w3s, 8);

    // max tiles = N/16 + (N_SPECIES-1 partial) -> 4100; 4 waves per block
    const int maxTiles = N_ATOMS / 16 + N_SPECIES;
    const int blocks   = (maxTiles + 3) / 4;
    const size_t ldsBytes = 4 * (8192 + 4096) * sizeof(uint16_t);  // 96 KB
    mlp_kernel<<<blocks, 128, ldsBytes, stream>>>(
        features, structIx, b1, b2, b3, W4, b4,
        counts, hdr, perm, w1s, w2s, w3s, out);
}
